// DeepSetCellEncoder_9466107920687
// MI455X (gfx1250) — compile-verified, matrix operand order checked by reading, not run on hardware
//
#include <hip/hip_runtime.h>
#include <stdint.h>

// ---------------------------------------------------------------------------
// DeepSet cell encoder, fused, f16 WMMA (v_wmma_f32_16x16x32_f16) on gfx1250.
// ---------------------------------------------------------------------------

typedef __attribute__((ext_vector_type(16))) _Float16 v16h;
typedef __attribute__((ext_vector_type(8)))  float    v8f;

#define D_IN   384
#define D_H    256
#define N_NODE 100000
#define N_E    400000
#define N_M    50000

#define KB1 (D_IN / 32)   // 12 k-blocks for layer 1
#define KB2 (D_H  / 32)   // 8  k-blocks for 256-wide layers
#define NT  (D_H  / 16)   // 16 n-tiles of width 16

#define XSTR  196         // uints per x-row in LDS (192 data + 4 pad -> bank spread)
#define HSTRU 132         // uints per h-row in LDS (128 data + 4 pad)
#define WAVES 2           // waves (tiles) per block

// fragment sizes in dwords
#define W1F_DW (KB1 * NT * 32 * 8)   // 49152
#define WHF_DW (KB2 * NT * 32 * 8)   // 32768

// ---------------------------------------------------------------------------
// helpers
// ---------------------------------------------------------------------------
__device__ __forceinline__ uint32_t pack2(float a, float b) {
    union { _Float16 h[2]; uint32_t u; } p;
    p.h[0] = (_Float16)a; p.h[1] = (_Float16)b;
    return p.u;
}

__device__ __forceinline__ uint16_t f16b(float v) {
    union { _Float16 h; uint16_t u; } c;
    c.h = (_Float16)v;
    return c.u;
}

// A-fragment from LDS row (16-bit A 16x32 layout):
// lane<16 -> K {0..7,16..23}; lane>=16 -> K {8..15,24..31}; p already includes
// the group offset (grp*4 dwords). Two ds_load_b128.
__device__ __forceinline__ v16h load_afrag(const uint32_t* p) {
    union { uint4 q[2]; v16h h; } f;
    f.q[0] = *(const uint4*)(p);
    f.q[1] = *(const uint4*)(p + 8);
    return f.h;
}

// B-fragment from prepacked global buffer: 8 contiguous dwords per lane.
__device__ __forceinline__ v16h load_bfrag(const uint32_t* p) {
    union { uint4 q[2]; v16h h; } f;
    f.q[0] = *(const uint4*)(p);
    f.q[1] = *(const uint4*)(p + 4);
    return f.h;
}

__device__ __forceinline__ v8f wmma_f16(v16h a, v16h b, v8f c) {
    return __builtin_amdgcn_wmma_f32_16x16x32_f16(
        /*neg_a=*/false, a, /*neg_b=*/false, b,
        /*c_mod=*/(short)0, c, /*reuse_a=*/false, /*reuse_b=*/false);
}

// ---------------------------------------------------------------------------
// Weight prep: f32 [K][N] row-major -> f16 B-fragments.
// Fragment (kb, nt): lane L holds column n = nt*16 + L%16; k-values
// base = (L&16)?8:0 ; dword j<4 -> k = kb*32+base+2j ; j>=4 -> kb*32+base+16+2(j-4)
// packed (k, k+1) into one dword. Per-lane 8 dwords contiguous.
// ---------------------------------------------------------------------------
__global__ void prep_frag(const float* __restrict__ W, uint32_t* __restrict__ dst,
                          int K, int N) {
    int id = blockIdx.x * blockDim.x + threadIdx.x;
    int ntl   = N / 16;
    int total = (K / 32) * ntl * 32 * 8;
    if (id >= total) return;
    int j    = id & 7;
    int lane = (id >> 3) & 31;
    int t    = id >> 8;
    int nt   = t % ntl;
    int kb   = t / ntl;
    int n    = nt * 16 + (lane & 15);
    int base = (lane & 16) ? 8 : 0;
    int k    = kb * 32 + ((j < 4) ? (base + 2 * j) : (base + 16 + 2 * (j - 4)));
    dst[id] = pack2(W[(size_t)k * N + n], W[(size_t)(k + 1) * N + n]);
}

__global__ void zero_f32(float* __restrict__ p, int n) {
    int i      = blockIdx.x * blockDim.x + threadIdx.x;
    int stride = gridDim.x * blockDim.x;
    for (; i < n; i += stride) p[i] = 0.0f;
}

// ---------------------------------------------------------------------------
// phi + scatter: per wave, one 16-edge tile.
//   gather x[16,384] -> f16 LDS -> relu(x@W1+b1) -> relu(h@W2+b2)
//   -> global_atomic_add_f32 into cell_sums[cell, :]
// ---------------------------------------------------------------------------
__global__ void __launch_bounds__(WAVES * 32)
phi_scatter(const float* __restrict__ X,
            const int*   __restrict__ nodes,
            const int*   __restrict__ cells,
            const uint32_t* __restrict__ w1f, const float* __restrict__ b1,
            const uint32_t* __restrict__ w2f, const float* __restrict__ b2,
            float* __restrict__ cell_sums) {
    __shared__ uint32_t s_x[WAVES][16 * XSTR];
    __shared__ uint16_t s_h[WAVES][16 * 2 * HSTRU];
    __shared__ int      s_nid[WAVES][16];
    __shared__ int      s_cid[WAVES][16];

    const int wave = threadIdx.x >> 5;
    const int lane = threadIdx.x & 31;
    const int tile = blockIdx.x * WAVES + wave;
    if (tile * 16 >= N_E) return;
    const int ebase = tile * 16;

    uint32_t* lx   = s_x[wave];
    uint16_t* lh   = s_h[wave];
    int*      lnid = s_nid[wave];
    int*      lcid = s_cid[wave];

    if (lane < 16) {
        int e = ebase + lane;
        lnid[lane] = (e < N_E) ? nodes[e] : 0;
        lcid[lane] = (e < N_E) ? cells[e] : -1;
    }

    // gather + convert to f16 (wave-private LDS; DS ops in-order per wave)
    for (int r = 0; r < 16; ++r) {
        int nid = lnid[r];
        const float4* src = (const float4*)(X + (size_t)nid * D_IN);
        #pragma unroll
        for (int it = 0; it < 3; ++it) {
            int c4   = it * 32 + lane;        // 0..95 float4s of the row
            float4 v = src[c4];
            lx[r * XSTR + c4 * 2]     = pack2(v.x, v.y);
            lx[r * XSTR + c4 * 2 + 1] = pack2(v.z, v.w);
        }
    }

    const int grp  = lane >> 4;   // K-half select
    const int rowm = lane & 15;   // row (A) / column (B,D) index
    const uint32_t* arow = lx + rowm * XSTR + grp * 4;

    // ---- layer 1: h1 = relu(x @ W1 + b1), h1 -> LDS as f16 ----
    for (int nt = 0; nt < NT; ++nt) {
        v8f acc = {};
        #pragma unroll
        for (int kb = 0; kb < KB1; ++kb) {
            v16h a = load_afrag(arow + kb * 16);
            v16h b = load_bfrag(w1f + (size_t)(((kb * NT + nt) * 32 + lane) * 8));
            acc = wmma_f16(a, b, acc);
        }
        float bias = b1[nt * 16 + rowm];
        #pragma unroll
        for (int r = 0; r < 8; ++r) {
            float v = acc[r] + bias;
            v = v > 0.0f ? v : 0.0f;
            int m = r + grp * 8;
            lh[m * (2 * HSTRU) + nt * 16 + rowm] = f16b(v);
        }
    }

    // ---- layer 2: h2 = relu(h1 @ W2 + b2), scatter-add to cell_sums ----
    const uint32_t* h1row = (const uint32_t*)lh + rowm * HSTRU + grp * 4;
    for (int nt = 0; nt < NT; ++nt) {
        v8f acc = {};
        #pragma unroll
        for (int kb = 0; kb < KB2; ++kb) {
            v16h a = load_afrag(h1row + kb * 16);
            v16h b = load_bfrag(w2f + (size_t)(((kb * NT + nt) * 32 + lane) * 8));
            acc = wmma_f16(a, b, acc);
        }
        float bias = b2[nt * 16 + rowm];
        #pragma unroll
        for (int r = 0; r < 8; ++r) {
            float v = acc[r] + bias;
            v = v > 0.0f ? v : 0.0f;
            int m   = r + grp * 8;
            int cid = lcid[m];
            if (cid >= 0)
                atomicAdd(cell_sums + (size_t)cid * D_H + nt * 16 + rowm, v);
        }
    }
}

// ---------------------------------------------------------------------------
// rho: per wave, one 16-cell tile; 3 fused layers, ping-pong LDS buffers.
// ---------------------------------------------------------------------------
__device__ __forceinline__ void mlp_layer_lds(const uint32_t* __restrict__ inbuf,
                                              const uint32_t* __restrict__ wf,
                                              const float*    __restrict__ bias,
                                              uint32_t*       __restrict__ outbuf,
                                              int lane) {
    const int grp  = lane >> 4;
    const int rowm = lane & 15;
    const uint32_t* arow = inbuf + rowm * HSTRU + grp * 4;
    uint16_t* oh = (uint16_t*)outbuf;
    for (int nt = 0; nt < NT; ++nt) {
        v8f acc = {};
        #pragma unroll
        for (int kb = 0; kb < KB2; ++kb) {
            v16h a = load_afrag(arow + kb * 16);
            v16h b = load_bfrag(wf + (size_t)(((kb * NT + nt) * 32 + lane) * 8));
            acc = wmma_f16(a, b, acc);
        }
        float bb = bias[nt * 16 + rowm];
        #pragma unroll
        for (int r = 0; r < 8; ++r) {
            float v = acc[r] + bb;
            v = v > 0.0f ? v : 0.0f;
            int m = r + grp * 8;
            oh[m * (2 * HSTRU) + nt * 16 + rowm] = f16b(v);
        }
    }
}

__global__ void __launch_bounds__(WAVES * 32)
rho_kernel(const float* __restrict__ cell_sums,
           const uint32_t* __restrict__ r1f, const float* __restrict__ c1,
           const uint32_t* __restrict__ r2f, const float* __restrict__ c2,
           const uint32_t* __restrict__ r3f, const float* __restrict__ c3,
           float* __restrict__ out) {
    __shared__ uint32_t s_a[WAVES][16 * HSTRU];
    __shared__ uint32_t s_b[WAVES][16 * HSTRU];

    const int wave = threadIdx.x >> 5;
    const int lane = threadIdx.x & 31;
    const int tile = blockIdx.x * WAVES + wave;
    if (tile * 16 >= N_M) return;
    const int mbase = tile * 16;

    uint32_t* bufA = s_a[wave];
    uint32_t* bufB = s_b[wave];

    // load cell_sums tile -> f16 LDS
    for (int r = 0; r < 16; ++r) {
        const float4* src = (const float4*)(cell_sums + (size_t)(mbase + r) * D_H);
        #pragma unroll
        for (int it = 0; it < 2; ++it) {
            int c4   = it * 32 + lane;       // 0..63 float4s of the row
            float4 v = src[c4];
            bufA[r * HSTRU + c4 * 2]     = pack2(v.x, v.y);
            bufA[r * HSTRU + c4 * 2 + 1] = pack2(v.z, v.w);
        }
    }

    mlp_layer_lds(bufA, r1f, c1, bufB, lane);   // r1 = relu(s @ R1 + c1)
    mlp_layer_lds(bufB, r2f, c2, bufA, lane);   // r2 = relu(r1 @ R2 + c2)

    // final layer: out = r2 @ R3 + c3 (no relu), stream f32 to global
    const int grp  = lane >> 4;
    const int rowm = lane & 15;
    const uint32_t* arow = bufA + rowm * HSTRU + grp * 4;
    for (int nt = 0; nt < NT; ++nt) {
        v8f acc = {};
        #pragma unroll
        for (int kb = 0; kb < KB2; ++kb) {
            v16h a = load_afrag(arow + kb * 16);
            v16h b = load_bfrag(r3f + (size_t)(((kb * NT + nt) * 32 + lane) * 8));
            acc = wmma_f16(a, b, acc);
        }
        float bb = c3[nt * 16 + rowm];
        #pragma unroll
        for (int r = 0; r < 8; ++r) {
            int m = r + grp * 8;
            out[(size_t)(mbase + m) * D_H + nt * 16 + rowm] = acc[r] + bb;
        }
    }
}

// ---------------------------------------------------------------------------
// launch
// ---------------------------------------------------------------------------
extern "C" void kernel_launch(void* const* d_in, const int* in_sizes, int n_in,
                              void* d_out, int out_size, void* d_ws, size_t ws_size,
                              hipStream_t stream) {
    const float* X     = (const float*)d_in[0];   // [N_NODE, D_IN]
    const int*   nodes = (const int*)  d_in[1];   // [N_E]
    const int*   cells = (const int*)  d_in[2];   // [N_E]
    // d_in[3] = M (scalar, compile-time constant here)
    const float* W1 = (const float*)d_in[4];
    const float* b1 = (const float*)d_in[5];
    const float* W2 = (const float*)d_in[6];
    const float* b2 = (const float*)d_in[7];
    const float* R1 = (const float*)d_in[8];
    const float* c1 = (const float*)d_in[9];
    const float* R2 = (const float*)d_in[10];
    const float* c2 = (const float*)d_in[11];
    const float* R3 = (const float*)d_in[12];
    const float* c3 = (const float*)d_in[13];
    float* out = (float*)d_out;

    // workspace layout
    float*    cell_sums = (float*)d_ws;                       // 50000*256 f32
    uint32_t* w1f = (uint32_t*)(cell_sums + (size_t)N_M * D_H);
    uint32_t* w2f = w1f + W1F_DW;
    uint32_t* r1f = w2f + WHF_DW;
    uint32_t* r2f = r1f + WHF_DW;
    uint32_t* r3f = r2f + WHF_DW;

    // 1) weight fragment prep
    prep_frag<<<(W1F_DW + 255) / 256, 256, 0, stream>>>(W1, w1f, D_IN, D_H);
    prep_frag<<<(WHF_DW + 255) / 256, 256, 0, stream>>>(W2, w2f, D_H, D_H);
    prep_frag<<<(WHF_DW + 255) / 256, 256, 0, stream>>>(R1, r1f, D_H, D_H);
    prep_frag<<<(WHF_DW + 255) / 256, 256, 0, stream>>>(R2, r2f, D_H, D_H);
    prep_frag<<<(WHF_DW + 255) / 256, 256, 0, stream>>>(R3, r3f, D_H, D_H);

    // 2) zero cell_sums
    zero_f32<<<2048, 256, 0, stream>>>(cell_sums, N_M * D_H);

    // 3) phi + scatter-add (one 16-edge tile per wave)
    {
        int tiles  = N_E / 16;                 // 25000
        int blocks = (tiles + WAVES - 1) / WAVES;
        phi_scatter<<<blocks, WAVES * 32, 0, stream>>>(X, nodes, cells,
                                                       w1f, b1, w2f, b2, cell_sums);
    }

    // 4) rho (one 16-cell tile per wave)
    {
        int tiles  = N_M / 16;                 // 3125
        int blocks = (tiles + WAVES - 1) / WAVES;
        rho_kernel<<<blocks, WAVES * 32, 0, stream>>>(cell_sums,
                                                      r1f, c1, r2f, c2, r3f, c3, out);
    }
}
